// MoE_8581344658167
// MI455X (gfx1250) — compile-verified, matrix operand order checked by reading, not run on hardware
//
#include <hip/hip_runtime.h>
#include <hip/hip_bf16.h>
#include <math.h>

// Problem constants (from reference): B=8, N=1024, E=1024, H=1536, X=8, K=2
#define Eq 1024
#define Hq 1536
#define Xq 8
#define Tq 8192           // B*N tokens
#define LDA 40            // LDS row stride in halves (32 + 8 pad; 80B = 16B aligned, kills bank conflicts)

typedef _Float16 half8  __attribute__((ext_vector_type(8)));
typedef _Float16 half16 __attribute__((ext_vector_type(16)));
typedef float    float8 __attribute__((ext_vector_type(8)));

// Load a 16x32 fp16 WMMA fragment (A: rows=M, B: rows=N — source must be [row][K] contiguous).
// ISA layout (CDNA5 ISA): lanes 0-15 hold row=lane, K-halves {0..7,16..23};
// lanes 16-31 hold row=lane-16, K-halves {8..15,24..31}.
__device__ __forceinline__ half16 frag_ld(const _Float16* __restrict__ base, int ld, int lane) {
    const int r  = lane & 15;
    const int kb = (lane & 16) ? 8 : 0;
    const _Float16* p = base + (size_t)r * ld + kb;
    half8 lo = *(const half8*)(p);
    half8 hi = *(const half8*)(p + 16);
    return __builtin_shufflevector(lo, hi, 0,1,2,3,4,5,6,7,8,9,10,11,12,13,14,15);
}

__device__ __forceinline__ half16 frag_ld_g(const _Float16* __restrict__ p) {
    half8 lo = *(const half8*)(p);
    half8 hi = *(const half8*)(p + 16);
    return __builtin_shufflevector(lo, hi, 0,1,2,3,4,5,6,7,8,9,10,11,12,13,14,15);
}

__device__ __forceinline__ float gelu_erf(float v) {
    return 0.5f * v * (1.0f + erff(v * 0.70710678118654752440f));
}

// Async copy of one 128-row x 32-half (64B) fp16 weight tile into LDS, [N][K] layout.
// 512 x 16B chunks over 256 threads -> 2 global_load_async_to_lds_b128 per thread (ASYNCcnt +2 per wave).
__device__ __forceinline__ void stage_async(const _Float16* __restrict__ wsrc, int row_stride,
                                            int n0, int k0, unsigned lds_base, int tid) {
#pragma unroll
    for (int j = 0; j < 2; ++j) {
        const int chunk = tid + 256 * j;          // 0..511
        const int row   = chunk >> 2;             // 0..127
        const int seg   = chunk & 3;              // 16B segment within the 64B row
        const _Float16* g = wsrc + (size_t)(n0 + row) * row_stride + k0 + seg * 8;
        const unsigned  l = lds_base + (unsigned)(row * LDA + seg * 8) * 2u;
        asm volatile("global_load_async_to_lds_b128 %0, %1, off" :: "v"(l), "v"(g) : "memory");
    }
}

// One 32-K GEMM step: 2 A-frags (global fp16) x 4 B-frags (LDS) -> 8 WMMAs.
__device__ __forceinline__ void gemm_step(const _Float16* __restrict__ bs,
                                          const _Float16* __restrict__ arow0,
                                          const _Float16* __restrict__ arow1,
                                          int k0, int waveN, int lane, float8 acc[2][4]) {
    half16 afr0 = frag_ld_g(arow0 + k0);
    half16 afr1 = frag_ld_g(arow1 + k0);
#pragma unroll
    for (int ni = 0; ni < 4; ++ni) {
        half16 bfr = frag_ld(bs + (waveN + ni * 16) * LDA, LDA, lane);
        acc[0][ni] = __builtin_amdgcn_wmma_f32_16x16x32_f16(false, afr0, false, bfr, (short)0, acc[0][ni], false, false);
        acc[1][ni] = __builtin_amdgcn_wmma_f32_16x16x32_f16(false, afr1, false, bfr, (short)0, acc[1][ni], false, false);
    }
}

// ---------------------------------------------------------------- zero output + counters
__global__ __launch_bounds__(256) void zero_kernel(float* __restrict__ out, int n4, int* __restrict__ counts) {
    int i = blockIdx.x * 256 + threadIdx.x;
    if (blockIdx.x == 0 && threadIdx.x < Xq) counts[threadIdx.x] = 0;
    if (i < n4) {
        float4 z; z.x = 0.f; z.y = 0.f; z.z = 0.f; z.w = 0.f;
        ((float4*)out)[i] = z;
    }
}

// ---------------------------------------------------------------- x (f32) -> xh (f16), straight copy
__global__ __launch_bounds__(256) void convert_x_kernel(const float* __restrict__ x, _Float16* __restrict__ xh) {
    const int i = blockIdx.x * 256 + threadIdx.x;       // one half8 per thread
    float4 a = ((const float4*)x)[i * 2 + 0];
    float4 b = ((const float4*)x)[i * 2 + 1];
    half8 h;
    h[0] = (_Float16)a.x; h[1] = (_Float16)a.y; h[2] = (_Float16)a.z; h[3] = (_Float16)a.w;
    h[4] = (_Float16)b.x; h[5] = (_Float16)b.y; h[6] = (_Float16)b.z; h[7] = (_Float16)b.w;
    ((half8*)xh)[i] = h;
}

// ---------------------------------------------------------------- per-expert transpose: src[x][R][C] f32 -> dst[x][C][R] f16
__global__ __launch_bounds__(256) void transpose_kernel(const float* __restrict__ src, _Float16* __restrict__ dst,
                                                        int R, int C) {
    __shared__ float tile[32][33];
    const int xex = blockIdx.z;
    const int r0 = blockIdx.x * 32, c0 = blockIdx.y * 32;
    const float* s = src + (size_t)xex * R * C;
    _Float16*    d = dst + (size_t)xex * C * R;

    const int tr = threadIdx.x >> 3;            // 0..31
    const int tc = (threadIdx.x & 7) * 4;       // 0..28
    float4 v = *(const float4*)(s + (size_t)(r0 + tr) * C + c0 + tc);
    tile[tr][tc + 0] = v.x; tile[tr][tc + 1] = v.y; tile[tr][tc + 2] = v.z; tile[tr][tc + 3] = v.w;
    __syncthreads();

    _Float16* dp = d + (size_t)(c0 + tr) * R + r0 + tc;
    dp[0] = (_Float16)tile[tc + 0][tr];
    dp[1] = (_Float16)tile[tc + 1][tr];
    dp[2] = (_Float16)tile[tc + 2][tr];
    dp[3] = (_Float16)tile[tc + 3][tr];
}

// ---------------------------------------------------------------- router: logits + top-2 + expert lists
__global__ __launch_bounds__(256) void router_kernel(const float* __restrict__ x,
                                                     const float* __restrict__ Wr,
                                                     const float* __restrict__ br,
                                                     float* __restrict__ topk_out,
                                                     int* __restrict__ counts,
                                                     int* __restrict__ lists) {
    const int lane = threadIdx.x & 31;
    const int wid  = threadIdx.x >> 5;
    const int t    = blockIdx.x * 8 + wid;   // one wave per token

    float s[Xq];
#pragma unroll
    for (int j = 0; j < Xq; ++j) s[j] = 0.f;

    const float* xr = x + (size_t)t * Eq;
    for (int e = lane; e < Eq; e += 32) {
        float xv = xr[e];
        const float4* w = (const float4*)(Wr + (size_t)e * Xq);
        float4 w0 = w[0], w1 = w[1];
        s[0] += xv * w0.x; s[1] += xv * w0.y; s[2] += xv * w0.z; s[3] += xv * w0.w;
        s[4] += xv * w1.x; s[5] += xv * w1.y; s[6] += xv * w1.z; s[7] += xv * w1.w;
    }
#pragma unroll
    for (int off = 16; off >= 1; off >>= 1)
#pragma unroll
        for (int j = 0; j < Xq; ++j) s[j] += __shfl_xor(s[j], off, 32);

    if (lane == 0) {
#pragma unroll
        for (int j = 0; j < Xq; ++j) s[j] += br[j];
        // softmax is monotonic -> top-2 of logits == top-2 of softmax scores.
        int i0 = 0; float b0 = s[0];
#pragma unroll
        for (int j = 1; j < Xq; ++j) if (s[j] > b0) { b0 = s[j]; i0 = j; }   // strict > : first index wins ties
        int i1 = (i0 == 0) ? 1 : 0; float b1v = s[i1];
#pragma unroll
        for (int j = 0; j < Xq; ++j) if (j != i0 && s[j] > b1v) { b1v = s[j]; i1 = j; }

        topk_out[t * 2 + 0] = (float)i0;
        topk_out[t * 2 + 1] = (float)i1;

        int s0 = atomicAdd(&counts[i0], 1); lists[i0 * Tq + s0] = t * 2 + 0;
        int s1 = atomicAdd(&counts[i1], 1); lists[i1 * Tq + s1] = t * 2 + 1;
    }
}

// ---------------------------------------------------------------- GEMM1: h = gelu(x_gathered @ W1[x] + b1[x]) -> fp16
// Block: M=128 tokens x N=128 H-cols, K-step 32. 8 waves as 4(M)x2(N); wave = 2 M-subtiles x 4 N-subtiles.
// A fragments direct from global fp16 (xh, gathered rows); B via 2x-unrolled double-buffered async-to-LDS.
__global__ __launch_bounds__(256) void gemm1_kernel(const _Float16* __restrict__ xh,
                                                    const _Float16* __restrict__ W1t,
                                                    const float* __restrict__ b1,
                                                    const int* __restrict__ counts,
                                                    const int* __restrict__ lists,
                                                    _Float16* __restrict__ hbuf) {
    const int xex = blockIdx.z;
    const int cnt = counts[xex];
    const int m0  = blockIdx.x * 128;
    if (m0 >= cnt) return;
    const int n0 = blockIdx.y * 128;

    __shared__ _Float16 B_s[2][128][LDA];
    __shared__ int toks_s[128];

    const int tid   = threadIdx.x;
    const int lane  = tid & 31;
    const int wid   = tid >> 5;
    const int waveM = (wid & 3) * 32;
    const int waveN = (wid >> 2) * 64;

    if (tid < 128) {
        int s = m0 + tid;
        if (s >= cnt) s = cnt - 1;           // clamp; clamped rows computed but never stored
        toks_s[tid] = lists[xex * Tq + s];   // value = token*2 + k  (== hbuf row id)
    }
    __syncthreads();

    const _Float16* wbase = W1t + (size_t)xex * Hq * Eq;   // [H][E] fp16
    const unsigned lb0 = (unsigned)(uintptr_t)&B_s[0][0][0];  // low 32 bits of generic == LDS offset
    const unsigned lb1 = (unsigned)(uintptr_t)&B_s[1][0][0];
    const _Float16* bs0 = &B_s[0][0][0];
    const _Float16* bs1 = &B_s[1][0][0];

    const int fr  = lane & 15;
    const int fkb = (lane & 16) ? 8 : 0;
    const _Float16* arow0 = xh + (size_t)(toks_s[waveM + fr]      >> 1) * Eq + fkb;
    const _Float16* arow1 = xh + (size_t)(toks_s[waveM + 16 + fr] >> 1) * Eq + fkb;

    float8 acc[2][4];
#pragma unroll
    for (int mi = 0; mi < 2; ++mi)
#pragma unroll
        for (int ni = 0; ni < 4; ++ni)
#pragma unroll
            for (int i = 0; i < 8; ++i) acc[mi][ni][i] = 0.f;

    const int NK = Eq / 32;   // 32 (even)
    stage_async(wbase, Eq, n0, 0, lb0, tid);

    for (int s = 0; s < NK; s += 2) {
        // prefetch stage s+1 into buf1 (always valid: NK even), compute buf0
        stage_async(wbase, Eq, n0, (s + 1) * 32, lb1, tid);
        asm volatile("s_wait_asynccnt 0x2" ::: "memory");   // stage s complete (in-order per wave)
        __syncthreads();
        gemm_step(bs0, arow0, arow1, s * 32, waveN, lane, acc);
        __syncthreads();                                    // buf0 free for overwrite

        // prefetch stage s+2 into buf0 (if any), compute buf1
        if (s + 2 < NK) {
            stage_async(wbase, Eq, n0, (s + 2) * 32, lb0, tid);
            asm volatile("s_wait_asynccnt 0x2" ::: "memory");
        } else {
            asm volatile("s_wait_asynccnt 0x0" ::: "memory");
        }
        __syncthreads();
        gemm_step(bs1, arow0, arow1, (s + 1) * 32, waveN, lane, acc);
        __syncthreads();                                    // buf1 free for overwrite
    }

    // epilogue: bias + exact-erf GELU, store fp16 h rows
    const int ncol = lane & 15;
    const int rsel = (lane >> 4) * 8;
#pragma unroll
    for (int ni = 0; ni < 4; ++ni) {
        const int n = n0 + waveN + ni * 16 + ncol;
        const float bias = b1[xex * Hq + n];
#pragma unroll
        for (int mi = 0; mi < 2; ++mi)
#pragma unroll
            for (int i = 0; i < 8; ++i) {
                const int m = waveM + mi * 16 + rsel + i;
                if (m0 + m < cnt) {
                    float v = acc[mi][ni][i] + bias;
                    hbuf[(size_t)toks_s[m] * Hq + n] = (_Float16)gelu_erf(v);
                }
            }
    }
}

// ---------------------------------------------------------------- GEMM2: out += 0.5*(h @ W2[x] + b2[x])
__global__ __launch_bounds__(256) void gemm2_kernel(const _Float16* __restrict__ hbuf,
                                                    const _Float16* __restrict__ W2t,
                                                    const float* __restrict__ b2,
                                                    const int* __restrict__ counts,
                                                    const int* __restrict__ lists,
                                                    float* __restrict__ out) {
    const int xex = blockIdx.z;
    const int cnt = counts[xex];
    const int m0  = blockIdx.x * 128;
    if (m0 >= cnt) return;
    const int n0 = blockIdx.y * 128;

    __shared__ _Float16 B_s[2][128][LDA];
    __shared__ int toks_s[128];

    const int tid   = threadIdx.x;
    const int lane  = tid & 31;
    const int wid   = tid >> 5;
    const int waveM = (wid & 3) * 32;
    const int waveN = (wid >> 2) * 64;

    if (tid < 128) {
        int s = m0 + tid;
        if (s >= cnt) s = cnt - 1;
        toks_s[tid] = lists[xex * Tq + s];
    }
    __syncthreads();

    const _Float16* wbase = W2t + (size_t)xex * Eq * Hq;   // [E][H] fp16
    const unsigned lb0 = (unsigned)(uintptr_t)&B_s[0][0][0];
    const unsigned lb1 = (unsigned)(uintptr_t)&B_s[1][0][0];
    const _Float16* bs0 = &B_s[0][0][0];
    const _Float16* bs1 = &B_s[1][0][0];

    const int fr  = lane & 15;
    const int fkb = (lane & 16) ? 8 : 0;
    const _Float16* arow0 = hbuf + (size_t)toks_s[waveM + fr]      * Hq + fkb;
    const _Float16* arow1 = hbuf + (size_t)toks_s[waveM + 16 + fr] * Hq + fkb;

    float8 acc[2][4];
#pragma unroll
    for (int mi = 0; mi < 2; ++mi)
#pragma unroll
        for (int ni = 0; ni < 4; ++ni)
#pragma unroll
            for (int i = 0; i < 8; ++i) acc[mi][ni][i] = 0.f;

    const int NK = Hq / 32;   // 48 (even)
    stage_async(wbase, Hq, n0, 0, lb0, tid);

    for (int s = 0; s < NK; s += 2) {
        stage_async(wbase, Hq, n0, (s + 1) * 32, lb1, tid);
        asm volatile("s_wait_asynccnt 0x2" ::: "memory");
        __syncthreads();
        gemm_step(bs0, arow0, arow1, s * 32, waveN, lane, acc);
        __syncthreads();

        if (s + 2 < NK) {
            stage_async(wbase, Hq, n0, (s + 2) * 32, lb0, tid);
            asm volatile("s_wait_asynccnt 0x2" ::: "memory");
        } else {
            asm volatile("s_wait_asynccnt 0x0" ::: "memory");
        }
        __syncthreads();
        gemm_step(bs1, arow0, arow1, (s + 1) * 32, waveN, lane, acc);
        __syncthreads();
    }

    // epilogue: out[token] += 0.5*(acc + b2)   (exactly two commutative float atomics per element -> deterministic)
    const int ncol = lane & 15;
    const int rsel = (lane >> 4) * 8;
#pragma unroll
    for (int ni = 0; ni < 4; ++ni) {
        const int n = n0 + waveN + ni * 16 + ncol;
        const float bias = b2[xex * Eq + n];
#pragma unroll
        for (int mi = 0; mi < 2; ++mi)
#pragma unroll
            for (int i = 0; i < 8; ++i) {
                const int m = waveM + mi * 16 + rsel + i;
                if (m0 + m < cnt) {
                    const int t = toks_s[m] >> 1;
                    atomicAdd(out + (size_t)t * Eq + n, 0.5f * (acc[mi][ni][i] + bias));
                }
            }
    }
}

// ---------------------------------------------------------------- launch
extern "C" void kernel_launch(void* const* d_in, const int* in_sizes, int n_in,
                              void* d_out, int out_size, void* d_ws, size_t ws_size,
                              hipStream_t stream) {
    (void)in_sizes; (void)n_in; (void)out_size; (void)ws_size;

    const float* x  = (const float*)d_in[0];
    const float* Wr = (const float*)d_in[1];
    const float* br = (const float*)d_in[2];
    const float* W1 = (const float*)d_in[3];
    const float* b1 = (const float*)d_in[4];
    const float* W2 = (const float*)d_in[5];
    const float* b2 = (const float*)d_in[6];

    float* out      = (float*)d_out;
    float* topk_out = out + (size_t)Tq * Eq;   // topk indices stored as floats after the main output

    // workspace layout (all 16B-aligned)
    char* ws = (char*)d_ws;
    int* counts    = (int*)ws;                                   // 8 ints
    int* lists     = (int*)(ws + 256);                           // X*T ints          = 256 KB
    size_t off     = 256 + (size_t)Xq * Tq * 4;
    _Float16* xh   = (_Float16*)(ws + off);  off += (size_t)Tq * Eq * 2;        // 16 MB
    _Float16* W1t  = (_Float16*)(ws + off);  off += (size_t)Xq * Hq * Eq * 2;   // 24 MB
    _Float16* W2t  = (_Float16*)(ws + off);  off += (size_t)Xq * Eq * Hq * 2;   // 24 MB
    _Float16* hbuf = (_Float16*)(ws + off);                                     // 2T x H = 48 MB

    zero_kernel     <<<(Tq * Eq / 4 + 255) / 256, 256, 0, stream>>>(out, Tq * Eq / 4, counts);
    convert_x_kernel<<<Tq * Eq / 8 / 256, 256, 0, stream>>>(x, xh);
    transpose_kernel<<<dim3(Eq / 32, Hq / 32, Xq), 256, 0, stream>>>(W1, W1t, Eq, Hq);
    transpose_kernel<<<dim3(Hq / 32, Eq / 32, Xq), 256, 0, stream>>>(W2, W2t, Hq, Eq);
    router_kernel   <<<Tq / 8, 256, 0, stream>>>(x, Wr, br, topk_out, counts, lists);
    gemm1_kernel    <<<dim3(Tq / 128, Hq / 128, Xq), 256, 0, stream>>>(xh, W1t, b1, counts, lists, hbuf);
    gemm2_kernel    <<<dim3(Tq / 128, Eq / 128, Xq), 256, 0, stream>>>(hbuf, W2t, b2, counts, lists, out);
}
